// QuadraticAttention_32452772889003
// MI455X (gfx1250) — compile-verified, matrix operand order checked by reading, not run on hardware
//
#include <hip/hip_runtime.h>

typedef __bf16  v16bf __attribute__((ext_vector_type(16)));
typedef float   v8f   __attribute__((ext_vector_type(8)));
typedef unsigned int v8u __attribute__((ext_vector_type(8)));
typedef unsigned int u32x4 __attribute__((ext_vector_type(4)));
typedef int i32x8 __attribute__((ext_vector_type(8)));
typedef int i32x4 __attribute__((ext_vector_type(4)));

#define Bsz 4
#define Nn  4096
#define Dd  256
#define MT  128     // output rows per workgroup
#define LT  64      // l-tile (double buffered)
#define STRD 264    // padded LDS row stride (elems): 128 DW + 4 DW pad
#define STRL 72     // padded LDS row stride (elems): 32 DW + 4 DW pad

__device__ __forceinline__ unsigned short f2bf(float f){
  unsigned int u = __float_as_uint(f);
  u += 0x7FFFu + ((u >> 16) & 1u);          // round-to-nearest-even
  return (unsigned short)(u >> 16);
}

// ---- CDNA5 async global->LDS copy (ASYNCcnt path) ----
__device__ __forceinline__ void async_cp16(unsigned lds, const unsigned short* g){
  unsigned long long ga = (unsigned long long)(uintptr_t)g;
  asm volatile("global_load_async_to_lds_b128 %0, %1, off"
               :: "v"(lds), "v"(ga) : "memory");
}
__device__ __forceinline__ void wait_async0(){
  asm volatile("s_wait_asynccnt 0x0" ::: "memory");
}
__device__ __forceinline__ unsigned lds_addr(const void* p){
  return (unsigned)(uintptr_t)p;   // low 32 bits of flat LDS address == LDS offset
}

// ---- CDNA5 Tensor Data Mover: 2D tile load, data_size=4B, LDS row padding ----
// tile_dw0  : dwords per tile row      tile_rows : tile rows
// ten_dw0   : tensor row length (dw)   ten_rows  : tensor rows (OOB bound)
// stride_dw : tensor row stride (dw)
// pad_k     : pad interval code (2^(k+1) dwords)   pad_a : pad amount code (a+1 dwords)
__device__ __forceinline__ void tdm_load_2d(unsigned lds, const void* gaddr,
    unsigned tile_dw0, unsigned tile_rows,
    unsigned ten_dw0, unsigned ten_rows, unsigned stride_dw,
    unsigned pad_k, unsigned pad_a){
  unsigned long long ga = (unsigned long long)(uintptr_t)gaddr;
  u32x4 g0;
  g0[0] = 1u;                                            // count=1, user descriptor
  g0[1] = lds;                                           // lds_addr
  g0[2] = (unsigned)(ga & 0xFFFFFFFFu);                  // global_addr[31:0]
  g0[3] = (unsigned)((ga >> 32) & 0x01FFFFFFu) | 0x80000000u; // global_addr[56:32], type=2
  i32x8 g1;
  g1[0] = (int)((2u << 16) | (1u << 20) | (pad_k << 22) | (pad_a << 25)); // data_size=4B, pad_en
  g1[1] = (int)((ten_dw0 & 0xFFFFu) << 16);              // tensor_dim0[15:0]
  g1[2] = (int)(((ten_dw0 >> 16) & 0xFFFFu) | ((ten_rows & 0xFFFFu) << 16));
  g1[3] = (int)(((ten_rows >> 16) & 0xFFFFu) | ((tile_dw0 & 0xFFFFu) << 16));
  g1[4] = (int)(tile_rows & 0xFFFFu);                    // tile_dim1; tile_dim2=0
  g1[5] = (int)stride_dw;                                // tensor_dim0_stride[31:0]
  g1[6] = 0;
  g1[7] = 0;
  i32x4 z4 = {0,0,0,0};
  i32x8 z8 = {0,0,0,0,0,0,0,0};
  __builtin_amdgcn_tensor_load_to_lds(g0, g1, z4, z4, z8, 0);
}
__device__ __forceinline__ void wait_tensor0(){
  __builtin_amdgcn_s_wait_tensorcnt((short)0);
}

// Load a 16x32 bf16 A/B fragment from row-major LDS.
__device__ __forceinline__ v16bf ldfrag(const unsigned short* base, int stride,
                                        int row0, int kbase, int lane){
  int row = row0 + (lane & 15);
  int k   = kbase + ((lane >> 4) << 3);
  const unsigned int* q  = (const unsigned int*)(base + row*stride + k);
  const unsigned int* q2 = (const unsigned int*)(base + row*stride + k + 16);
  v8u t;
  t[0]=q[0];  t[1]=q[1];  t[2]=q[2];  t[3]=q[3];
  t[4]=q2[0]; t[5]=q2[1]; t[6]=q2[2]; t[7]=q2[3];
  return __builtin_bit_cast(v16bf, t);
}

// ---- prep: one-time f32 -> bf16 conversion (x row-major, x transposed, W) ----
__global__ __launch_bounds__(256)
void qa_prep(const float* __restrict__ x, const float* __restrict__ W,
             unsigned short* __restrict__ xbf, unsigned short* __restrict__ xTbf,
             unsigned short* __restrict__ wbf){
  const int idx = blockIdx.x * 256 + threadIdx.x;       // quad index
  const int b   = idx >> 18;                            // / (Nn*Dd/4)
  const int rem = idx & ((Nn*Dd/4) - 1);
  const int n   = rem >> 6;                             // / (Dd/4)
  const int dq  = (rem & 63) << 2;
  const float4 v = *(const float4*)(x + ((size_t)b*Nn + n)*Dd + dq);
  unsigned short h0 = f2bf(v.x), h1 = f2bf(v.y), h2 = f2bf(v.z), h3 = f2bf(v.w);
  uint2 pk; pk.x = (unsigned)h0 | ((unsigned)h1 << 16);
            pk.y = (unsigned)h2 | ((unsigned)h3 << 16);
  *(uint2*)&xbf[((size_t)b*Nn + n)*Dd + dq] = pk;
  xTbf[((size_t)b*Dd + dq+0)*Nn + n] = h0;
  xTbf[((size_t)b*Dd + dq+1)*Nn + n] = h1;
  xTbf[((size_t)b*Dd + dq+2)*Nn + n] = h2;
  xTbf[((size_t)b*Dd + dq+3)*Nn + n] = h3;
  if (idx < Nn*Dd/4){
    const int wn = idx >> 6, wdq = (idx & 63) << 2;
    const float4 wv = *(const float4*)(W + (size_t)wn*Dd + wdq);
    uint2 wp; wp.x = (unsigned)f2bf(wv.x) | ((unsigned)f2bf(wv.y) << 16);
              wp.y = (unsigned)f2bf(wv.z) | ((unsigned)f2bf(wv.w) << 16);
    *(uint2*)&wbf[(size_t)wn*Dd + wdq] = wp;
  }
}

__global__ __launch_bounds__(256)
void qa_fused(const unsigned short* __restrict__ xbf,
              const unsigned short* __restrict__ xTbf,
              const unsigned short* __restrict__ wbf,
              float* __restrict__ out){
  __shared__ __align__(16) unsigned short sXr[MT*STRD];       // X rows (bf16)
  __shared__ __align__(16) unsigned short sW [2][LT*STRD];    // W tile, double buffered
  __shared__ __align__(16) unsigned short sXT[2][Dd*STRL];    // X^T tile, double buffered
  __shared__ __align__(16) unsigned short sS [MT*STRL];       // S tile

  const int tid  = threadIdx.x;
  const int lane = tid & 31;
  const int wave = tid >> 5;
  const int wr   = wave >> 1;   // 0..3 : 32-row M super-block
  const int wc   = wave & 1;    // 0..1 : half of S cols / half of D cols

  const int tiles = Nn / MT;    // 32
  const int b  = blockIdx.x / tiles;
  const int n0 = (blockIdx.x % tiles) * MT;

  const unsigned short* xb  = xbf  + (size_t)b * Nn * Dd;
  const unsigned short* xTb = xTbf + (size_t)b * Dd * Nn;

  // ---- prologue: async-stage X rows; TDM-stage tile 0 ----
#pragma unroll
  for (int c = 0; c < 16; ++c){                 // sXr: 128 rows x 32 chunks (ASYNCcnt path)
    int chunk = c*256 + tid;
    int row = chunk >> 5, off = (chunk & 31) << 3;
    async_cp16(lds_addr(&sXr[row*STRD + off]), xb + (size_t)(n0+row)*Dd + off);
  }
  if (wave == 0){
    tdm_load_2d(lds_addr(&sW[0][0]),  wbf, 128,  64, 128, Nn, 128, 6, 3);
    tdm_load_2d(lds_addr(&sXT[0][0]), xTb,  32, 256, Nn/2, Dd, Nn/2, 4, 3);
  }
  wait_async0();
  if (wave == 0) wait_tensor0();
  __syncthreads();

  v8f outAcc[2][8];
#pragma unroll
  for (int mi = 0; mi < 2; ++mi)
#pragma unroll
    for (int t = 0; t < 8; ++t) outAcc[mi][t] = (v8f)0.0f;

  int p = 0;
  for (int l0 = 0; l0 < Nn; l0 += LT){
    // ---- TDM: stream NEXT tile into the other buffer while we compute ----
    if (l0 + LT < Nn && wave == 0){
      const int ln = l0 + LT, q = p ^ 1;
      tdm_load_2d(lds_addr(&sW[q][0]),  wbf + (size_t)ln*Dd, 128,  64, 128, Nn, 128, 6, 3);
      tdm_load_2d(lds_addr(&sXT[q][0]), xTb + ln,             32, 256, Nn/2, Dd, Nn/2, 4, 3);
    }

    // ---- GEMM1: S[i,j] = sum_d Xr[i,d] * W[l0+j,d] ----
#pragma unroll
    for (int mi = 0; mi < 2; ++mi){
      const int mrow = wr*32 + mi*16;
      v8f sacc[2];
#pragma unroll
      for (int jt = 0; jt < 2; ++jt) sacc[jt] = (v8f)0.0f;
#pragma unroll
      for (int k0 = 0; k0 < Dd; k0 += 32){
        v16bf a = ldfrag(sXr, STRD, mrow, k0, lane);
#pragma unroll
        for (int jt = 0; jt < 2; ++jt){
          v16bf bf = ldfrag(sW[p], STRD, wc*32 + jt*16, k0, lane);
          sacc[jt] = __builtin_amdgcn_wmma_f32_16x16x32_bf16(
              false, a, false, bf, (short)0, sacc[jt], false, false);
        }
      }
#pragma unroll
      for (int jt = 0; jt < 2; ++jt){
        int col   = wc*32 + jt*16 + (lane & 15);
        int rbase = mrow + ((lane >> 4) << 3);
#pragma unroll
        for (int v = 0; v < 8; ++v)
          sS[(rbase + v)*STRL + col] = f2bf(sacc[jt][v]);
      }
    }
    __syncthreads();   // sS visible to all waves

    // ---- GEMM2: out[i,d] += sum_j S[i,j] * X[l0+j,d] ----
#pragma unroll
    for (int k0 = 0; k0 < LT; k0 += 32){
      v16bf a0 = ldfrag(sS, STRL, wr*32,      k0, lane);
      v16bf a1 = ldfrag(sS, STRL, wr*32 + 16, k0, lane);
#pragma unroll
      for (int t = 0; t < 8; ++t){
        v16bf bf = ldfrag(sXT[p], STRL, wc*128 + t*16, k0, lane);
        outAcc[0][t] = __builtin_amdgcn_wmma_f32_16x16x32_bf16(
            false, a0, false, bf, (short)0, outAcc[0][t], false, false);
        outAcc[1][t] = __builtin_amdgcn_wmma_f32_16x16x32_bf16(
            false, a1, false, bf, (short)0, outAcc[1][t], false, false);
      }
    }

    if (wave == 0) wait_tensor0();   // next tile landed in the other buffer
    __syncthreads();                 // everyone done reading sW[p]/sXT[p]/sS
    p ^= 1;
  }

  // ---- epilogue: C layout -> global f32 ----
#pragma unroll
  for (int mi = 0; mi < 2; ++mi){
    const int m0 = n0 + wr*32 + mi*16 + ((lane >> 4) << 3);
#pragma unroll
    for (int t = 0; t < 8; ++t){
      int d = wc*128 + t*16 + (lane & 15);
#pragma unroll
      for (int v = 0; v < 8; ++v)
        out[((size_t)b*Nn + m0 + v)*Dd + d] = outAcc[mi][t][v];
    }
  }
}

extern "C" void kernel_launch(void* const* d_in, const int* in_sizes, int n_in,
                              void* d_out, int out_size, void* d_ws, size_t ws_size,
                              hipStream_t stream) {
  (void)in_sizes; (void)n_in; (void)out_size; (void)ws_size;
  const float* x = (const float*)d_in[0];
  const float* W = (const float*)d_in[1];
  float* out = (float*)d_out;

  unsigned short* xbf  = (unsigned short*)d_ws;                 // [B][N][D]
  unsigned short* xTbf = xbf  + (size_t)Bsz*Nn*Dd;              // [B][D][N]
  unsigned short* wbf  = xTbf + (size_t)Bsz*Nn*Dd;              // [N][D]

  qa_prep<<<dim3(Bsz*Nn*Dd/4/256), 256, 0, stream>>>(x, W, xbf, xTbf, wbf);
  qa_fused<<<dim3(Bsz*(Nn/MT)), 256, 0, stream>>>(xbf, xTbf, wbf, out);
}
